// SNRModule_85280870630034
// MI455X (gfx1250) — matrix operand order repaired and checked by reading, not compiled, not run on hardware
//
#include <hip/hip_runtime.h>
#include <hip/hip_bf16.h>
#include <cfloat>

#define NODES   100000
#define HID     256
#define EDGES   3200000
#define OUTF    2
#define SLOPE   0.2f

typedef __attribute__((ext_vector_type(16))) _Float16 v16h;
typedef __attribute__((ext_vector_type(8)))  float    v8f;

// Order-preserving float <-> signed-int encoding (for atomicMax on floats).
__device__ __forceinline__ int enc_f32(float f) {
    int i = __float_as_int(f);
    return (i >= 0) ? i : ((~i) ^ 0x80000000);
}
__device__ __forceinline__ float dec_f32(int k) {
    int i = (k >= 0) ? k : (~(k ^ 0x80000000));
    return __int_as_float(i);
}

// ---------------------------------------------------------------- init
__global__ void k_init(int* __restrict__ m_enc, float* __restrict__ denom,
                       float* __restrict__ num0, float* __restrict__ num1) {
    int i = blockIdx.x * blockDim.x + threadIdx.x;
    if (i < NODES) {
        m_enc[i] = enc_f32(-FLT_MAX);
        denom[i] = 0.0f;
        num0[i]  = 0.0f;
        num1[i]  = 0.0f;
    }
}

// ---------------------------------------------------- z = input @ W (WMMA)
// One wave = one 16-row tile, K = 256 as 8 chunks of 32 (f16, f32 accum).
__global__ void k_proj_wmma(const float* __restrict__ inp,
                            const float* __restrict__ W,
                            const float* __restrict__ attn_l,
                            const float* __restrict__ attn_r,
                            float* __restrict__ z,
                            float* __restrict__ el,
                            float* __restrict__ er) {
    const int lane     = threadIdx.x & 31;
    const int wave     = threadIdx.x >> 5;
    const int row_base = blockIdx.x * 128 + wave * 16;   // 8 waves * 16 rows
    if (row_base >= NODES) return;                       // wave-uniform; EXEC stays full

    const int r   = row_base + (lane & 15);  // A-matrix row (M)
    const int grp = lane >> 4;               // half-wave group
    const int col = lane & 15;               // B/C/D column (N)

    // B = W (HIDxOUTF) zero-padded to 16 columns.
    // 32x16 f16 B: elems 0..7 = K = c*32 + grp*8 + j, elems 8..15 = K = c*32 + 16 + grp*8 + j.
    v16h bmat[8];
    #pragma unroll
    for (int c = 0; c < 8; ++c) {
        const int kc = c * 32 + grp * 8;
        #pragma unroll
        for (int j = 0; j < 16; ++j) {
            float w = (col < OUTF) ? W[(kc + (j < 8 ? j : j + 8)) * OUTF + col] : 0.0f;
            bmat[c][j] = (_Float16)w;
        }
    }

    // A: 16x32 f16 per chunk. Lane holds two 8-long runs:
    //   elems 0..7  = K = kc + grp*8 + j
    //   elems 8..15 = K = kc + 16 + grp*8 + j
    v8f acc = {};
    const float* rowp = inp + (size_t)r * HID;
    #pragma unroll
    for (int c = 0; c < 8; ++c) {
        const int k0 = c * 32 + grp * 8;
        float4 f0 = *(const float4*)(rowp + k0);
        float4 f1 = *(const float4*)(rowp + k0 + 4);
        float4 f2 = *(const float4*)(rowp + k0 + 16);
        float4 f3 = *(const float4*)(rowp + k0 + 20);
        v16h a;
        a[0]  = (_Float16)f0.x;  a[1]  = (_Float16)f0.y;
        a[2]  = (_Float16)f0.z;  a[3]  = (_Float16)f0.w;
        a[4]  = (_Float16)f1.x;  a[5]  = (_Float16)f1.y;
        a[6]  = (_Float16)f1.z;  a[7]  = (_Float16)f1.w;
        a[8]  = (_Float16)f2.x;  a[9]  = (_Float16)f2.y;
        a[10] = (_Float16)f2.z;  a[11] = (_Float16)f2.w;
        a[12] = (_Float16)f3.x;  a[13] = (_Float16)f3.y;
        a[14] = (_Float16)f3.z;  a[15] = (_Float16)f3.w;
        acc = __builtin_amdgcn_wmma_f32_16x16x32_f16(
                  false, a, false, bmat[c], (short)0, acc, false, false);
    }

    // C/D layout: VGPR v, lanes 0-15 -> M=v, lanes 16-31 -> M=v+8; N=lane&15.
    const float al0 = attn_l[0], al1 = attn_l[1];
    const float ar0 = attn_r[0], ar1 = attn_r[1];
    #pragma unroll
    for (int v = 0; v < 8; ++v) {
        const int row = row_base + v + grp * 8;
        float d0 = acc[v];
        float d1 = __shfl_xor(d0, 1, 32);   // partner column (0<->1)
        if (col < OUTF) z[row * OUTF + col] = d0;
        if (col == 0) {
            el[row] = d0 * al0 + d1 * al1;
            er[row] = d0 * ar0 + d1 * ar1;
        }
    }
}

// ------------------------------------------------- edge pass 1: leaky + max
__global__ void k_edge_max(const int* __restrict__ esrc, const int* __restrict__ edst,
                           const float* __restrict__ el, const float* __restrict__ er,
                           float* __restrict__ ebuf, int* __restrict__ m_enc) {
    int t = blockIdx.x * blockDim.x + threadIdx.x;
    if (t >= EDGES) return;
    int s = esrc[t], d = edst[t];
    float e = el[s] + er[d];
    e = (e > 0.0f) ? e : SLOPE * e;
    ebuf[t] = e;
    atomicMax(&m_enc[d], enc_f32(e));
}

// -------------------------------------- edge pass 2: exp + fused num/denom
__global__ void k_edge_acc(const int* __restrict__ esrc, const int* __restrict__ edst,
                           const float* __restrict__ ebuf, const int* __restrict__ m_enc,
                           const float* __restrict__ z,
                           float* __restrict__ denom,
                           float* __restrict__ num0, float* __restrict__ num1) {
    int t = blockIdx.x * blockDim.x + threadIdx.x;
    if (t >= EDGES) return;
    int s = esrc[t], d = edst[t];
    float m  = dec_f32(m_enc[d]);
    float ex = __expf(ebuf[t] - m);
    atomicAdd(&denom[d], ex);
    atomicAdd(&num0[d], ex * z[s * OUTF + 0]);
    atomicAdd(&num1[d], ex * z[s * OUTF + 1]);
}

// ------------------------------------------- per-node sigmoid gate scalar
__global__ void k_node_scale(const float* __restrict__ x, const float* __restrict__ bias,
                             const float* __restrict__ denom,
                             const float* __restrict__ num0, const float* __restrict__ num1,
                             float* __restrict__ sbuf) {
    int i = blockIdx.x * blockDim.x + threadIdx.x;
    if (i >= NODES) return;
    float den  = denom[i];
    float inv  = (den > 0.0f) ? (1.0f / den) : 0.0f;
    float c0   = num0[i] * inv + bias[0];
    float c1   = num1[i] * inv + bias[1];
    float stdv = fmaxf(c0, 0.0f);
    float mnv  = fmaxf(c1, 0.0f);
    float a    = x[i] * stdv + mnv;
    sbuf[i] = 1.0f / (1.0f + __expf(-a));
}

// --------------------------------------------- out = input * s[row] (f4)
__global__ void k_scale_out(const float4* __restrict__ inp,
                            const float* __restrict__ sbuf,
                            float4* __restrict__ out) {
    const int total = NODES * (HID / 4);
    int idx = blockIdx.x * blockDim.x + threadIdx.x;
    if (idx >= total) return;
    if (idx + 2048 < total) __builtin_prefetch(inp + idx + 2048, 0, 1);
    float s  = sbuf[idx >> 6];          // HID/4 = 64 float4 per row
    float4 v = inp[idx];
    v.x *= s; v.y *= s; v.z *= s; v.w *= s;
    out[idx] = v;
}

extern "C" void kernel_launch(void* const* d_in, const int* in_sizes, int n_in,
                              void* d_out, int out_size, void* d_ws, size_t ws_size,
                              hipStream_t stream) {
    (void)in_sizes; (void)n_in; (void)out_size; (void)ws_size;

    const float* inp    = (const float*)d_in[0];   // [N, H]
    const float* x      = (const float*)d_in[1];   // [N, 1]
    /* d_in[2] = degree (unused by forward) */
    const int*   esrc   = (const int*)d_in[3];     // [E]
    const int*   edst   = (const int*)d_in[4];     // [E]
    const float* W      = (const float*)d_in[5];   // [H, 2]
    const float* attn_l = (const float*)d_in[6];   // [2]
    const float* attn_r = (const float*)d_in[7];   // [2]
    const float* bias   = (const float*)d_in[8];   // [2]
    float*       out    = (float*)d_out;           // [N, H]

    // Workspace layout (all 4-byte aligned).
    char* ws = (char*)d_ws;
    float* z     = (float*)ws; ws += (size_t)NODES * OUTF * sizeof(float);
    float* el    = (float*)ws; ws += (size_t)NODES * sizeof(float);
    float* er    = (float*)ws; ws += (size_t)NODES * sizeof(float);
    int*   m_enc = (int*)  ws; ws += (size_t)NODES * sizeof(int);
    float* denom = (float*)ws; ws += (size_t)NODES * sizeof(float);
    float* num0  = (float*)ws; ws += (size_t)NODES * sizeof(float);
    float* num1  = (float*)ws; ws += (size_t)NODES * sizeof(float);
    float* sbuf  = (float*)ws; ws += (size_t)NODES * sizeof(float);
    float* ebuf  = (float*)ws; ws += (size_t)EDGES * sizeof(float);

    const int BT = 256;
    k_init<<<(NODES + BT - 1) / BT, BT, 0, stream>>>(m_enc, denom, num0, num1);
    k_proj_wmma<<<(NODES + 127) / 128, BT, 0, stream>>>(inp, W, attn_l, attn_r, z, el, er);
    k_edge_max<<<(EDGES + BT - 1) / BT, BT, 0, stream>>>(esrc, edst, el, er, ebuf, m_enc);
    k_edge_acc<<<(EDGES + BT - 1) / BT, BT, 0, stream>>>(esrc, edst, ebuf, m_enc, z,
                                                         denom, num0, num1);
    k_node_scale<<<(NODES + BT - 1) / BT, BT, 0, stream>>>(x, bias, denom, num0, num1, sbuf);
    const int total4 = NODES * (HID / 4);
    k_scale_out<<<(total4 + BT - 1) / BT, BT, 0, stream>>>((const float4*)inp, sbuf,
                                                           (float4*)out);
}
